// CausalSelfAttention_34437047779548
// MI455X (gfx1250) — matrix operand order repaired and checked by reading, not compiled, not run on hardware
//
#include <hip/hip_runtime.h>
#include <hip/hip_bf16.h>

typedef _Float16 v16h __attribute__((ext_vector_type(16)));
typedef _Float16 v2h  __attribute__((ext_vector_type(2)));
typedef float    v8f  __attribute__((ext_vector_type(8)));
typedef float    v2f  __attribute__((ext_vector_type(2)));

#define Bc  2
#define Nc  2048
#define Ec  1024
#define Hc  16
#define HDc 64
#define F3  192   // 3*HD

__device__ __forceinline__ v8f wmma_f16(v16h a, v16h b, v8f c) {
    // D = A(16x32 f16) * B(32x16 f16) + C(16x16 f32)
    return __builtin_amdgcn_wmma_f32_16x16x32_f16(false, a, false, b, (short)0, c, false, false);
}

// ---------------- weight prep: transpose + fp32->fp16 ----------------
// WqkvT[(h*192+f)*E + e] = Wqkv[h][e][f]
__global__ void prep_wqkv_kernel(const float* __restrict__ Wqkv, _Float16* __restrict__ WqkvT) {
    int idx = blockIdx.x * blockDim.x + threadIdx.x;
    if (idx >= Hc * F3 * Ec) return;
    int e = idx % Ec;
    int r = idx / Ec;
    int f = r % F3;
    int h = r / F3;
    WqkvT[idx] = (_Float16)Wqkv[((size_t)h * Ec + e) * F3 + f];
}

// WoutT[e*E + k] = Wout[k][e]   (column-major Wout)
__global__ void prep_wout_kernel(const float* __restrict__ Wout, _Float16* __restrict__ WoutT) {
    int idx = blockIdx.x * blockDim.x + threadIdx.x;
    if (idx >= Ec * Ec) return;
    int k = idx % Ec;
    int e = idx / Ec;
    WoutT[idx] = (_Float16)Wout[(size_t)k * Ec + e];
}

// ---------------- kernel 1: per-head QKV projection (16 x 64 per wave) ----------------
// One wave computes a 16-row x 64-col block of proj[b,h] = x[b] @ Wqkv[h] + bqkv[h].
// The 64-col block is exactly one segment: seg 0=K, 1=Q (scaled 1/8), 2=V (stored transposed).
__global__ __launch_bounds__(32)
void qkv_kernel(const float* __restrict__ x, const _Float16* __restrict__ WqkvT,
                const float* __restrict__ bqkv,
                _Float16* __restrict__ Qh, _Float16* __restrict__ Kh, _Float16* __restrict__ Vt) {
    int blk = blockIdx.x;
    int seg = blk % 3;                   // 0=K, 1=Q, 2=V
    int nt  = (blk / 3) & 127;
    int h   = (blk / (3 * 128)) & 15;
    int b   = blk / (3 * 128 * 16);
    int lane = threadIdx.x & 31;
    int col = lane & 15, hi = lane >> 4;
    int m0 = nt * 16, f0 = seg * 64;

    const float* Arow = x + ((size_t)b * Nc + m0 + col) * Ec;           // A: row = lane&15
    const _Float16* Bcol[4];
#pragma unroll
    for (int t = 0; t < 4; ++t)
        Bcol[t] = WqkvT + ((size_t)h * F3 + f0 + 16 * t + col) * Ec;    // B: col = lane&15

    v8f zero = {0.f,0.f,0.f,0.f,0.f,0.f,0.f,0.f};
    v8f acc[4];
#pragma unroll
    for (int t = 0; t < 4; ++t) acc[t] = zero;

    for (int k0 = 0; k0 < Ec; k0 += 32) {
        v16h a;
#pragma unroll
        for (int d = 0; d < 8; ++d) {
            int ka = k0 + ((d >> 2) * 16 + hi * 8 + (d & 3) * 2);       // A-layout (16-bit 16x32)
            v2f av = *(const v2f*)(Arow + ka);
            a[2 * d]     = (_Float16)av.x;
            a[2 * d + 1] = (_Float16)av.y;
        }
#pragma unroll
        for (int t = 0; t < 4; ++t) {
            v16h bm;
#pragma unroll
            for (int d = 0; d < 8; ++d) {
                int kb = k0 + (d >> 2) * 16 + hi * 8 + (d & 3) * 2;     // B-layout (16-bit 32x16)
                v2h bv = *(const v2h*)(Bcol[t] + kb);
                bm[2 * d]     = bv.x;
                bm[2 * d + 1] = bv.y;
            }
            acc[t] = wmma_f16(a, bm, acc[t]);
        }
    }

    size_t bh = (size_t)(b * Hc + h);
#pragma unroll
    for (int t = 0; t < 4; ++t) {
        float bias = bqkv[h * F3 + f0 + 16 * t + col];
        int d0 = 16 * t + col;                                           // head-dim index
#pragma unroll
        for (int i = 0; i < 8; ++i) {
            int m = m0 + i + 8 * hi;  // D-layout: row = vgpr + 8*hi, col = lane&15
            float v = acc[t][i] + bias;
            if (seg == 0)      Kh[(bh * Nc + m) * HDc + d0] = (_Float16)v;
            else if (seg == 1) Qh[(bh * Nc + m) * HDc + d0] = (_Float16)(v * 0.125f);
            else               Vt[(bh * HDc + d0) * Nc + m] = (_Float16)v;   // transposed
        }
    }
}

// ---------------- kernel 2: flash causal attention, one wave per (b,h,16 q-rows) -------
__global__ __launch_bounds__(32)
void attn_kernel(const _Float16* __restrict__ Qh, const _Float16* __restrict__ Kh,
                 const _Float16* __restrict__ Vt, _Float16* __restrict__ Sbuf) {
    __shared__ _Float16 Plds[16 * 32];   // P tile staging: D-layout -> A-layout transpose

    int blk = blockIdx.x;
    int qt = blk & 127;
    int h  = (blk >> 7) & 15;
    int b  = blk >> 11;
    int lane = threadIdx.x & 31;
    int col = lane & 15, hi = lane >> 4;
    int q0 = qt * 16;
    size_t bh = (size_t)(b * Hc + h);

    // Preload Q A-fragments for hd=[0,32) and [32,64)
    const _Float16* Qrow = Qh + (bh * Nc + q0 + col) * HDc;
    v16h aq0, aq1;
#pragma unroll
    for (int d = 0; d < 8; ++d) {
        int ka = (d >> 2) * 16 + hi * 8 + (d & 3) * 2;
        v2h q0v = *(const v2h*)(Qrow + ka);
        aq0[2 * d] = q0v.x; aq0[2 * d + 1] = q0v.y;
        v2h q1v = *(const v2h*)(Qrow + 32 + ka);
        aq1[2 * d] = q1v.x; aq1[2 * d + 1] = q1v.y;
    }

    float rmax[8], rsum[8];
    v8f o[4];
    v8f zero = {0.f,0.f,0.f,0.f,0.f,0.f,0.f,0.f};
#pragma unroll
    for (int i = 0; i < 8; ++i) { rmax[i] = -1e30f; rsum[i] = 0.f; }
#pragma unroll
    for (int t = 0; t < 4; ++t) o[t] = zero;

    int jmax = (q0 + 15) / 32;   // last double-key-tile touching the causal region
    for (int j2 = 0; j2 <= jmax; ++j2) {
        // ---- scores for 2 key tiles of 16 (keys [32*j2, 32*j2+32)) ----
        v8f st[2];
#pragma unroll
        for (int t = 0; t < 2; ++t) {
            int key = 32 * j2 + 16 * t + col;           // B col = key index
            const _Float16* Krow = Kh + (bh * Nc + key) * HDc;
            v16h b0, b1;
#pragma unroll
            for (int d = 0; d < 8; ++d) {
                int kb = (d >> 2) * 16 + hi * 8 + (d & 3) * 2;
                v2h k0v = *(const v2h*)(Krow + kb);
                b0[2 * d] = k0v.x; b0[2 * d + 1] = k0v.y;
                v2h k1v = *(const v2h*)(Krow + 32 + kb);
                b1[2 * d] = k1v.x; b1[2 * d + 1] = k1v.y;
            }
            v8f s = zero;
            s = wmma_f16(aq0, b0, s);
            s = wmma_f16(aq1, b1, s);
#pragma unroll
            for (int i = 0; i < 8; ++i) {               // causal mask (no -inf => no NaN)
                int m = q0 + i + 8 * hi;
                if (key > m) s[i] = -1e30f;
            }
            st[t] = s;
        }

        // ---- online softmax update (row reductions across the 16-lane half-wave) ----
        float pr0[8], pr1[8];
#pragma unroll
        for (int i = 0; i < 8; ++i) {
            float mx = fmaxf(st[0][i], st[1][i]);
#pragma unroll
            for (int w = 1; w < 16; w <<= 1) mx = fmaxf(mx, __shfl_xor(mx, w, 32));
            float nm = fmaxf(rmax[i], mx);
            float corr = __expf(rmax[i] - nm);
            float p0 = __expf(st[0][i] - nm);
            float p1 = __expf(st[1][i] - nm);
            float ls = p0 + p1;
#pragma unroll
            for (int w = 1; w < 16; w <<= 1) ls += __shfl_xor(ls, w, 32);
            rsum[i] = rsum[i] * corr + ls;
            rmax[i] = nm;
#pragma unroll
            for (int t = 0; t < 4; ++t) o[t][i] *= corr;
            pr0[i] = p0; pr1[i] = p1;
        }

        // ---- transpose P (D-layout) into A-layout via LDS ----
#pragma unroll
        for (int i = 0; i < 8; ++i) {
            int m = i + 8 * hi;
            Plds[m * 32 + col]      = (_Float16)pr0[i];
            Plds[m * 32 + 16 + col] = (_Float16)pr1[i];
        }
        __syncthreads();
        v16h pa;
#pragma unroll
        for (int d = 0; d < 8; ++d) {
            int kl = (d >> 2) * 16 + hi * 8 + (d & 3) * 2;
            v2h pv = *(const v2h*)(&Plds[col * 32 + kl]);
            pa[2 * d] = pv.x; pa[2 * d + 1] = pv.y;
        }
        __syncthreads();

        // ---- O += P(16x32) x V(32x64): 4 WMMAs, contiguous loads from transposed V ----
#pragma unroll
        for (int t2 = 0; t2 < 4; ++t2) {
            const _Float16* Vrow = Vt + (bh * HDc + 16 * t2 + col) * Nc + 32 * j2;
            v16h bv;
#pragma unroll
            for (int d = 0; d < 8; ++d) {
                int kb = (d >> 2) * 16 + hi * 8 + (d & 3) * 2;
                v2h vv = *(const v2h*)(Vrow + kb);
                bv[2 * d] = vv.x; bv[2 * d + 1] = vv.y;
            }
            o[t2] = wmma_f16(pa, bv, o[t2]);
        }
    }

    // ---- normalize and write concat-head buffer [B,N,E] (f16) ----
#pragma unroll
    for (int i = 0; i < 8; ++i) {
        int m = q0 + i + 8 * hi;
        float inv = 1.0f / rsum[i];
        size_t base = ((size_t)b * Nc + m) * Ec + h * HDc;
#pragma unroll
        for (int t2 = 0; t2 < 4; ++t2)
            Sbuf[base + 16 * t2 + col] = (_Float16)(o[t2][i] * inv);
    }
}

// ---------------- kernel 3: output projection (16 x 64 per wave) ----------------
__global__ __launch_bounds__(32)
void outproj_kernel(const _Float16* __restrict__ Sbuf, const _Float16* __restrict__ WoutT,
                    const float* __restrict__ bout, float* __restrict__ out) {
    int blk = blockIdx.x;
    int et = blk & 15;        // 16 groups of 64 output cols
    int mt = blk >> 4;        // 256 row tiles of flattened [B*N, E]
    int lane = threadIdx.x & 31;
    int col = lane & 15, hi = lane >> 4;
    int m0 = mt * 16, e0 = et * 64;

    const _Float16* Arow = Sbuf + ((size_t)(m0 + col)) * Ec;
    const _Float16* Bcol[4];
#pragma unroll
    for (int t = 0; t < 4; ++t)
        Bcol[t] = WoutT + ((size_t)(e0 + 16 * t + col)) * Ec;

    v8f zero = {0.f,0.f,0.f,0.f,0.f,0.f,0.f,0.f};
    v8f acc[4];
#pragma unroll
    for (int t = 0; t < 4; ++t) acc[t] = zero;

    for (int k0 = 0; k0 < Ec; k0 += 32) {
        v16h a;
#pragma unroll
        for (int d = 0; d < 8; ++d) {
            int ka = k0 + (d >> 2) * 16 + hi * 8 + (d & 3) * 2;
            v2h av = *(const v2h*)(Arow + ka);
            a[2 * d] = av.x; a[2 * d + 1] = av.y;
        }
#pragma unroll
        for (int t = 0; t < 4; ++t) {
            v16h bm;
#pragma unroll
            for (int d = 0; d < 8; ++d) {
                int kb = k0 + (d >> 2) * 16 + hi * 8 + (d & 3) * 2;
                v2h bv = *(const v2h*)(Bcol[t] + kb);
                bm[2 * d] = bv.x; bm[2 * d + 1] = bv.y;
            }
            acc[t] = wmma_f16(a, bm, acc[t]);
        }
    }

#pragma unroll
    for (int t = 0; t < 4; ++t) {
        float bias = bout[e0 + 16 * t + col];
#pragma unroll
        for (int i = 0; i < 8; ++i) {
            int m = m0 + i + 8 * hi;
            out[(size_t)m * Ec + e0 + 16 * t + col] = acc[t][i] + bias;
        }
    }
}

// ---------------- launcher ----------------
extern "C" void kernel_launch(void* const* d_in, const int* in_sizes, int n_in,
                              void* d_out, int out_size, void* d_ws, size_t ws_size,
                              hipStream_t stream) {
    (void)in_sizes; (void)n_in; (void)out_size; (void)ws_size;
    const float* x    = (const float*)d_in[0];
    const float* Wqkv = (const float*)d_in[1];
    const float* bqkv = (const float*)d_in[2];
    const float* Wout = (const float*)d_in[3];
    const float* bout = (const float*)d_in[4];
    float* out = (float*)d_out;

    char* ws = (char*)d_ws;
    const size_t SZ_QKV = (size_t)Bc * Hc * Nc * HDc * sizeof(_Float16);  // 8 MB each
    _Float16* Qh    = (_Float16*)(ws);
    _Float16* Kh    = (_Float16*)(ws + SZ_QKV);
    _Float16* Vt    = (_Float16*)(ws + 2 * SZ_QKV);
    _Float16* Sbuf  = (_Float16*)(ws + 3 * SZ_QKV);                       // B*N*E f16 = 8 MB
    _Float16* WqkvT = (_Float16*)(ws + 4 * SZ_QKV);                       // 6 MB
    _Float16* WoutT = (_Float16*)(ws + 4 * SZ_QKV + (size_t)Hc * F3 * Ec * sizeof(_Float16)); // 2 MB

    prep_wqkv_kernel<<<(Hc * F3 * Ec + 255) / 256, 256, 0, stream>>>(Wqkv, WqkvT);
    prep_wout_kernel<<<(Ec * Ec + 255) / 256, 256, 0, stream>>>(Wout, WoutT);

    // B * H * (N/16) * (192/64) = 2*16*128*3 waves, each 16x64
    qkv_kernel<<<Bc * Hc * 128 * 3, 32, 0, stream>>>(x, WqkvT, bqkv, Qh, Kh, Vt);

    // B * H * (N/16) = 4096 waves
    attn_kernel<<<Bc * Hc * 128, 32, 0, stream>>>(Qh, Kh, Vt, Sbuf);

    // (B*N/16) * (E/64) = 256*16 waves, each 16x64
    outproj_kernel<<<256 * 16, 32, 0, stream>>>(Sbuf, WoutT, bout, out);
}
